// LSTM_60979945669191
// MI455X (gfx1250) — matrix operand order repaired and checked by reading, not compiled, not run on hardware
//
#include <hip/hip_runtime.h>
#include <hip/hip_bf16.h>

typedef __bf16 v16bf __attribute__((ext_vector_type(16)));
typedef float  v8f   __attribute__((ext_vector_type(8)));

namespace {
constexpr int kT = 512, kB = 256, kD = 256, kH = 256, kO = 64;
constexpr int kNBlk = 16;               // persistent workgroups (one 16-col H slice each)
constexpr int kHBufElems = kB * kH;     // 65536 bf16 per h buffer

// workspace byte offsets
constexpr size_t kOffW4T = 0;                       // 1024*512*2  = 1048576 B
constexpr size_t kOffWyT = 1048576;                 // 64*256*2    = 32768 B
constexpr size_t kOffHbf = kOffWyT + 32768;         // 2 buffers   = 262144 B
constexpr size_t kOffBar = kOffHbf + 262144;        // 2 ints

constexpr int kPrepW4T = 1024 * 512;
constexpr int kPrepWyT = 64 * 256;
constexpr int kPrepHbf = 2 * kHBufElems;
constexpr int kPrepTot = kPrepW4T + kPrepWyT + kPrepHbf + 2;
}

union AFrag {
  v16bf  v;
  __bf16 e[16];
  uint4  q[2];
};

__device__ __forceinline__ float fsig(float x)  { return 1.0f / (1.0f + __expf(-x)); }
__device__ __forceinline__ float ftanh(float x) { return 2.0f / (1.0f + __expf(-2.0f * x)) - 1.0f; }

// ---------------- preprocessing: transpose+convert weights, zero state ----------------
__global__ __launch_bounds__(256) void lstm_prep_kernel(
    const float* __restrict__ Wf, const float* __restrict__ Wi,
    const float* __restrict__ Wc, const float* __restrict__ Wo,
    const float* __restrict__ Wy,
    __bf16* __restrict__ w4t, __bf16* __restrict__ wyt,
    __bf16* __restrict__ hbf, int* __restrict__ bar)
{
  const int i = blockIdx.x * 256 + threadIdx.x;
  if (i < kPrepW4T) {                       // W4T[n][k] = Wg[k][j], n = g*256+j
    const int n = i >> 9, k = i & 511;
    const int g = n >> 8, jc = n & 255;
    const float* W = (g == 0) ? Wf : (g == 1) ? Wi : (g == 2) ? Wc : Wo;
    w4t[i] = (__bf16)W[k * 256 + jc];
  } else if (i < kPrepW4T + kPrepWyT) {     // WyT[o][k] = Wy[k][o]
    const int r = i - kPrepW4T;
    const int o = r >> 8, k = r & 255;
    wyt[r] = (__bf16)Wy[k * 64 + o];
  } else if (i < kPrepW4T + kPrepWyT + kPrepHbf) {
    hbf[i - (kPrepW4T + kPrepWyT)] = (__bf16)0.0f;   // h0 = 0, both buffers
  } else if (i < kPrepTot) {
    bar[i - (kPrepW4T + kPrepWyT + kPrepHbf)] = 0;   // barrier count/generation
  }
}

// ---------------- persistent LSTM time loop ----------------
__global__ __launch_bounds__(256) void lstm_persist_kernel(
    const float* __restrict__ x, const int* __restrict__ lens,
    const __bf16* __restrict__ w4t,
    const float* __restrict__ bfp, const float* __restrict__ bip,
    const float* __restrict__ bcp, const float* __restrict__ bop,
    __bf16* __restrict__ hbf, float* __restrict__ h_out,
    int* __restrict__ bar)
{
  // 64 KB: this block's weight slice, pre-swizzled into WMMA B-fragment order:
  // wl[((g*16 + kt)*32 + lane)*16 + e]
  __shared__ __bf16 wl[4 * 16 * 32 * 16];

  const int tid  = threadIdx.x;
  const int wv   = tid >> 5;
  const int lane = tid & 31;
  const int hf   = lane >> 4;       // K-half selector (WMMA lane grouping)
  const int jj   = lane & 15;       // output column within slice / A row within tile
  const int j0   = blockIdx.x * 16; // this block's H columns
  const int j    = j0 + jj;
  const int m0   = wv * 32;         // this wave's 32 batch rows

  for (int i = tid; i < 4 * 16 * 32 * 16; i += 256) {
    const int e  = i & 15;
    const int L  = (i >> 4) & 31;
    const int kt = (i >> 9) & 15;
    const int g  = i >> 13;
    const int n  = g * 256 + j0 + (L & 15);        // B column (gate-major)
    const int k  = kt * 32 + (e >> 3) * 16 + (L >> 4) * 8 + (e & 7);    // B row (K)
    wl[i] = w4t[n * 512 + k];
  }
  __syncthreads();

  const float bfv = bfp[j], biv = bip[j], bcv = bcp[j], bov = bop[j];

  int   len[16];
  float cst[16], hst[16];     // cell + hidden state live in registers for all 512 steps
#pragma unroll
  for (int r = 0; r < 2; ++r)
#pragma unroll
    for (int v = 0; v < 8; ++v) {
      const int m = m0 + r * 16 + v + 8 * hf;
      len[r * 8 + v] = lens[m];
      cst[r * 8 + v] = 0.0f;
      hst[r * 8 + v] = 0.0f;
    }

  for (int t = 0; t < kT; ++t) {
    const float*  xt  = x + (size_t)t * kB * kD;
    const __bf16* hrd = hbf + (size_t)(t & 1) * kHBufElems;
    __bf16*       hwr = hbf + (size_t)((t + 1) & 1) * kHBufElems;

    v8f acc[2][4];
    const v8f vz = {0.f, 0.f, 0.f, 0.f, 0.f, 0.f, 0.f, 0.f};
#pragma unroll
    for (int r = 0; r < 2; ++r)
#pragma unroll
      for (int g = 0; g < 4; ++g) acc[r][g] = vz;

    // ---- K = 0..255 : input part, fp32 x converted to bf16 on the fly ----
#pragma unroll
    for (int kt = 0; kt < 8; ++kt) {
      AFrag a[2];
#pragma unroll
      for (int r = 0; r < 2; ++r) {
        const float* px = xt + (size_t)(m0 + r * 16 + jj) * kD + kt * 32 + hf * 8;
        const float4 f0 = *(const float4*)(px);
        const float4 f1 = *(const float4*)(px + 4);
        const float4 f2 = *(const float4*)(px + 16);
        const float4 f3 = *(const float4*)(px + 20);
        const float tmp[16] = {f0.x, f0.y, f0.z, f0.w, f1.x, f1.y, f1.z, f1.w,
                               f2.x, f2.y, f2.z, f2.w, f3.x, f3.y, f3.z, f3.w};
#pragma unroll
        for (int e = 0; e < 16; ++e) a[r].e[e] = (__bf16)tmp[e];
      }
#pragma unroll
      for (int g = 0; g < 4; ++g) {
        const v16bf bmat = *(const v16bf*)(wl + ((g * 16 + kt) * 32 + lane) * 16);
        acc[0][g] = __builtin_amdgcn_wmma_f32_16x16x32_bf16(false, a[0].v, false, bmat, (short)0, acc[0][g], false, false);
        acc[1][g] = __builtin_amdgcn_wmma_f32_16x16x32_bf16(false, a[1].v, false, bmat, (short)0, acc[1][g], false, false);
      }
    }
    // ---- K = 256..511 : recurrent part, bf16 h from L2 ----
#pragma unroll
    for (int kt = 8; kt < 16; ++kt) {
      AFrag a[2];
#pragma unroll
      for (int r = 0; r < 2; ++r) {
        const __bf16* ph = hrd + (size_t)(m0 + r * 16 + jj) * kH + (kt - 8) * 32 + hf * 8;
        a[r].q[0] = *(const uint4*)(ph);
        a[r].q[1] = *(const uint4*)(ph + 16);
      }
#pragma unroll
      for (int g = 0; g < 4; ++g) {
        const v16bf bmat = *(const v16bf*)(wl + ((g * 16 + kt) * 32 + lane) * 16);
        acc[0][g] = __builtin_amdgcn_wmma_f32_16x16x32_bf16(false, a[0].v, false, bmat, (short)0, acc[0][g], false, false);
        acc[1][g] = __builtin_amdgcn_wmma_f32_16x16x32_bf16(false, a[1].v, false, bmat, (short)0, acc[1][g], false, false);
      }
    }

    // ---- gates + masked (packed-sequence) state update ----
#pragma unroll
    for (int r = 0; r < 2; ++r)
#pragma unroll
      for (int v = 0; v < 8; ++v) {
        const int s = r * 8 + v;
        const float fg = fsig (acc[r][0][v] + bfv);
        const float ig = fsig (acc[r][1][v] + biv);
        const float cg = ftanh(acc[r][2][v] + bcv);
        const float og = fsig (acc[r][3][v] + bov);
        const float cn = fg * cst[s] + ig * cg;
        const float hn = og * ftanh(cn);
        const bool  act = t < len[s];
        cst[s] = act ? cn : cst[s];
        const float hv = act ? hn : hst[s];
        hst[s] = hv;
        const int m = m0 + r * 16 + v + 8 * hf;
        hwr[(size_t)m * kH + j] = (__bf16)hv;   // unconditional: carries h for inactive rows
      }

    // prefetch next timestep's x rows for this wave (global_prefetch_b8)
    if (t + 1 < kT) {
      const float* xn = x + (size_t)(t + 1) * kB * kD + (size_t)(m0 + lane) * kD;
#pragma unroll
      for (int p = 0; p < 8; ++p) __builtin_prefetch(xn + p * 32, 0, 0);
    }

    // ---- device-wide generation barrier across the 16 persistent workgroups ----
    __threadfence();
    __syncthreads();
    if (tid == 0) {
      const int gen = __hip_atomic_load(&bar[1], __ATOMIC_RELAXED, __HIP_MEMORY_SCOPE_AGENT);
      const int n   = __hip_atomic_fetch_add(&bar[0], 1, __ATOMIC_ACQ_REL, __HIP_MEMORY_SCOPE_AGENT);
      if (n == kNBlk - 1) {
        __hip_atomic_store(&bar[0], 0, __ATOMIC_RELAXED, __HIP_MEMORY_SCOPE_AGENT);
        __hip_atomic_fetch_add(&bar[1], 1, __ATOMIC_RELEASE, __HIP_MEMORY_SCOPE_AGENT);
      } else {
        while (__hip_atomic_load(&bar[1], __ATOMIC_ACQUIRE, __HIP_MEMORY_SCOPE_AGENT) == gen)
          __builtin_amdgcn_s_sleep(1);
      }
    }
    __syncthreads();
    __threadfence();
  }

  // final fp32 hidden state -> second output section
#pragma unroll
  for (int r = 0; r < 2; ++r)
#pragma unroll
    for (int v = 0; v < 8; ++v) {
      const int m = m0 + r * 16 + v + 8 * hf;
      h_out[(size_t)m * kH + j] = hst[r * 8 + v];
    }
}

// ---------------- epilogue: y = h @ Wy + by (WMMA) ----------------
__global__ __launch_bounds__(256) void lstm_y_kernel(
    const __bf16* __restrict__ hbf, const __bf16* __restrict__ wyt,
    const float* __restrict__ by, float* __restrict__ y)
{
  const int tid = threadIdx.x, wv = tid >> 5, lane = tid & 31;
  const int hf = lane >> 4, jj = lane & 15;
  const int idx = blockIdx.x * 8 + wv;    // 64 tiles total
  const int rt = idx >> 2, ct = idx & 3;

  v8f acc = {0.f, 0.f, 0.f, 0.f, 0.f, 0.f, 0.f, 0.f};
#pragma unroll
  for (int kt = 0; kt < 8; ++kt) {
    AFrag a;
    const __bf16* ph = hbf + (size_t)(rt * 16 + jj) * kH + kt * 32 + hf * 16;
    a.q[0] = *(const uint4*)(ph);
    a.q[1] = *(const uint4*)(ph + 8);
    const v16bf bmat = *(const v16bf*)(wyt + (size_t)(ct * 16 + jj) * kH + kt * 32 + hf * 16);
    acc = __builtin_amdgcn_wmma_f32_16x16x32_bf16(false, a.v, false, bmat, (short)0, acc, false, false);
  }
  const int col = ct * 16 + jj;
  const float bias = by[col];
#pragma unroll
  for (int v = 0; v < 8; ++v) {
    const int m = rt * 16 + v + 8 * hf;
    y[m * kO + col] = acc[v] + bias;
  }
}

extern "C" void kernel_launch(void* const* d_in, const int* in_sizes, int n_in,
                              void* d_out, int out_size, void* d_ws, size_t ws_size,
                              hipStream_t stream) {
  (void)in_sizes; (void)n_in; (void)out_size; (void)ws_size;
  const float* x    = (const float*)d_in[0];
  const int*   lens = (const int*)d_in[1];
  const float* Wf   = (const float*)d_in[2];
  const float* bf_  = (const float*)d_in[3];
  const float* Wi   = (const float*)d_in[4];
  const float* bi_  = (const float*)d_in[5];
  const float* Wc   = (const float*)d_in[6];
  const float* bc_  = (const float*)d_in[7];
  const float* Wo   = (const float*)d_in[8];
  const float* bo_  = (const float*)d_in[9];
  const float* Wy   = (const float*)d_in[10];
  const float* by_  = (const float*)d_in[11];

  char* ws = (char*)d_ws;
  __bf16* w4t = (__bf16*)(ws + kOffW4T);
  __bf16* wyt = (__bf16*)(ws + kOffWyT);
  __bf16* hbf = (__bf16*)(ws + kOffHbf);
  int*    bar = (int*)   (ws + kOffBar);

  float* y_out = (float*)d_out;            // [256 x 64]
  float* h_out = y_out + kB * kO;          // [256 x 256]

  lstm_prep_kernel<<<(kPrepTot + 255) / 256, 256, 0, stream>>>(
      Wf, Wi, Wc, Wo, Wy, w4t, wyt, hbf, bar);
  lstm_persist_kernel<<<kNBlk, 256, 0, stream>>>(
      x, lens, w4t, bf_, bi_, bc_, bo_, hbf, h_out, bar);
  // T is even, so the final h lives in buffer 0 of the double buffer
  lstm_y_kernel<<<8, 256, 0, stream>>>(hbf, wyt, by_, y_out);
}
